// L1SSIMLoss_10582799417362
// MI455X (gfx1250) — matrix operand order repaired and checked by reading, not compiled, not run on hardware
//
#include <hip/hip_runtime.h>

typedef __attribute__((ext_vector_type(16))) _Float16 v16h;
typedef __attribute__((ext_vector_type(8)))  _Float16 v8h;
typedef __attribute__((ext_vector_type(2)))  _Float16 v2h;
typedef __attribute__((ext_vector_type(8)))  float    v8f;

union V16 { v16h v; v8h h[2]; _Float16 e[16]; };
union V8  { v8f  v; float e[8]; };

// 11-tap Gaussian, sigma=1.5, normalized (sum = 1)
__constant__ float G11[11] = {
    0.00102838f, 0.00759900f, 0.03600077f, 0.10936068f, 0.21300553f,
    0.26601172f, 0.21300553f, 0.10936068f, 0.03600077f, 0.00759900f,
    0.00102838f};

__device__ inline float gband(int d) {
  return (d >= 0 && d <= 10) ? G11[d] : 0.0f;
}

#define WMMA16(A, B, C) \
  __builtin_amdgcn_wmma_f32_16x16x32_f16(false, (A), false, (B), (short)0, (C), false, false)

constexpr int IMH = 512;
constexpr int IMW = 512;
constexpr int CST = 40;   // column stride in halfs (80B: 16B aligned, bank-spread)
constexpr int TOTAL_TILES = 48 * 32 * 32;

__global__ void zero_acc_kernel(float* acc) {
  acc[0] = 0.0f;
  acc[1] = 0.0f;
}

__global__ __launch_bounds__(256) void ssim_l1_kernel(const float* __restrict__ X,
                                                      const float* __restrict__ Y,
                                                      float* __restrict__ acc) {
  // per-wave private column-major f16 halos (8 waves per block)
  __shared__ _Float16 sX[8][32 * CST];
  __shared__ _Float16 sY[8][32 * CST];

  const int tid   = threadIdx.x;
  const int wave  = tid >> 5;
  const int lane  = tid & 31;
  const int halfw = lane >> 4;   // 0: lanes 0-15, 1: lanes 16-31
  const int mn    = lane & 15;   // M for A/C fragments, N for B/C fragments

  int tileId = blockIdx.x * 8 + wave;
  if (tileId >= TOTAL_TILES) tileId = TOTAL_TILES - 1;  // grid is exact; safety clamp
  const int plane = tileId >> 10;
  const int t     = tileId & 1023;
  const int tr    = (t >> 5) << 4;   // tile origin row
  const int tc    = (t & 31) << 4;   // tile origin col
  const float* Xp = X + (size_t)plane * (IMH * IMW);
  const float* Yp = Y + (size_t)plane * (IMH * IMW);

  _Float16* hx = sX[wave];
  _Float16* hy = sY[wave];

  // ---- Halo fill: coalesced rows, transposed (column-major) f16 store.
  // Valid region rows/cols 0..25 (global -5..+20 around tile); everything else zero.
  float l1_sum = 0.0f;
  const int  gc    = tc - 5 + lane;
  const int  gcc   = min(max(gc, 0), IMW - 1);
  const bool colok = (gc >= 0) && (gc < IMW) && (lane < 26);
#pragma unroll
  for (int i = 0; i < 26; i += 2) {
    const int gr0  = tr - 5 + i;
    const int gr1  = gr0 + 1;
    const int g0c  = min(max(gr0, 0), IMH - 1);
    const int g1c  = min(max(gr1, 0), IMH - 1);
    float x0 = Xp[g0c * IMW + gcc];
    float y0 = Yp[g0c * IMW + gcc];
    float x1 = Xp[g1c * IMW + gcc];
    float y1 = Yp[g1c * IMW + gcc];
    const bool ok0 = colok && (gr0 >= 0) && (gr0 < IMH);
    const bool ok1 = colok && (gr1 >= 0) && (gr1 < IMH);
    if (!ok0) { x0 = 0.0f; y0 = 0.0f; }
    if (!ok1) { x1 = 0.0f; y1 = 0.0f; }
    v2h px = {(_Float16)x0, (_Float16)x1};
    v2h py = {(_Float16)y0, (_Float16)y1};
    *(v2h*)&hx[lane * CST + i] = px;
    *(v2h*)&hy[lane * CST + i] = py;
    // exact-fp32 L1 over the 16x16 interior (rows/cols 5..20); always in-image there
    if (i >= 5 && i < 21 && lane >= 5 && lane < 21) l1_sum += fabsf(x0 - y0);
    if (i + 1 >= 5 && i + 1 < 21 && lane >= 5 && lane < 21) l1_sum += fabsf(x1 - y1);
  }
  {  // zero rows 26..31 of every column (keeps stale LDS NaN/Inf out of WMMA)
    const v2h zz = {(_Float16)0.0f, (_Float16)0.0f};
    *(v2h*)&hx[lane * CST + 26] = zz;
    *(v2h*)&hx[lane * CST + 28] = zz;
    *(v2h*)&hx[lane * CST + 30] = zz;
    *(v2h*)&hy[lane * CST + 26] = zz;
    *(v2h*)&hy[lane * CST + 28] = zz;
    *(v2h*)&hy[lane * CST + 30] = zz;
  }
  __syncthreads();

  // ---- Single constant banded-Gaussian B-fragment, shared by BOTH passes:
  // vertical:   V^T = Q^T x Gv^T,  Gv^T[j][m] = g[j-m]
  // horizontal: O   = V   x Gh,    Gh[k][n]   = g[k-n]
  V16 bW;
#pragma unroll
  for (int h = 0; h < 16; ++h) {
    const int kb = (h & 7) + 8 * halfw + 16 * (h >> 3);  // 16-bit B layout K index
    bW.e[h] = (_Float16)gband(kb - mn);
  }

  // ---- Data A-fragments (Q^T, 16-col chunks), loaded once as contiguous b128 runs.
  // A element h -> K(row) = (h&7) + 8*halfw + 16*(h>>3): two contiguous 8-half groups.
  V16 ax0, ax1, ay0, ay1;
  {
    const int r0 = 8 * halfw;
    ax0.h[0] = *(const v8h*)&hx[mn * CST + r0];
    ax0.h[1] = *(const v8h*)&hx[mn * CST + r0 + 16];
    ax1.h[0] = *(const v8h*)&hx[(mn + 16) * CST + r0];
    ax1.h[1] = *(const v8h*)&hx[(mn + 16) * CST + r0 + 16];
    ay0.h[0] = *(const v8h*)&hy[mn * CST + r0];
    ay0.h[1] = *(const v8h*)&hy[mn * CST + r0 + 16];
    ay1.h[0] = *(const v8h*)&hy[(mn + 16) * CST + r0];
    ay1.h[1] = *(const v8h*)&hy[(mn + 16) * CST + r0 + 16];
  }

  // ---- Separable conv via WMMA for x, y, x^2, y^2, xy — no LDS bounce between passes:
  // the C layout of V^T (lane = out-row m, VGPR = column) IS the A layout of pass 2.
  V8 res[5];
#pragma unroll
  for (int q = 0; q < 5; ++q) {
    V16 a0, a1;
    if (q == 0)      { a0 = ax0; a1 = ax1; }
    else if (q == 1) { a0 = ay0; a1 = ay1; }
    else if (q == 2) { a0.v = ax0.v * ax0.v; a1.v = ax1.v * ax1.v; }
    else if (q == 3) { a0.v = ay0.v * ay0.v; a1.v = ay1.v * ay1.v; }
    else             { a0.v = ax0.v * ay0.v; a1.v = ax1.v * ay1.v; }

    v8f z0 = {}, z1 = {};
    V8 c0, c1;
    c0.v = WMMA16(a0.v, bW.v, z0);   // V^T cols 0..15
    c1.v = WMMA16(a1.v, bW.v, z1);   // V^T cols 16..31

    V16 aV;                          // register-only relayout (f32 -> f16)
#pragma unroll
    for (int h = 0; h < 8; ++h) {
      aV.e[h]     = (_Float16)c0.e[h];
      aV.e[h + 8] = (_Float16)c1.e[h];
    }
    v8f z2 = {};
    res[q].v = WMMA16(aV.v, bW.v, z2);  // horizontal conv
  }

  // ---- Pointwise SSIM map (all five results share C layout; mean is layout-agnostic)
  const float C1c = 0.0001f;  // (0.01)^2
  const float C2c = 0.0009f;  // (0.03)^2
  float ssim_sum = 0.0f;
#pragma unroll
  for (int i = 0; i < 8; ++i) {
    const float mu1 = res[0].e[i];
    const float mu2 = res[1].e[i];
    const float m11 = mu1 * mu1;
    const float m22 = mu2 * mu2;
    const float m12 = mu1 * mu2;
    const float s1  = res[2].e[i] - m11;
    const float s2  = res[3].e[i] - m22;
    const float s12 = res[4].e[i] - m12;
    const float num = (2.0f * m12 + C1c) * (2.0f * s12 + C2c);
    const float den = (m11 + m22 + C1c) * (s1 + s2 + C2c);
    ssim_sum += num / den;
  }

  // ---- Wave32 reduction + global accumulation
#pragma unroll
  for (int off = 16; off > 0; off >>= 1) {
    ssim_sum += __shfl_xor(ssim_sum, off, 32);
    l1_sum   += __shfl_xor(l1_sum, off, 32);
  }
  if (lane == 0) {
    atomicAdd(&acc[0], ssim_sum);
    atomicAdd(&acc[1], l1_sum);
  }
}

__global__ void finalize_kernel(const float* __restrict__ acc, float* __restrict__ out) {
  const float inv = 1.0f / (16.0f * 3.0f * 512.0f * 512.0f);
  // L1_WEIGHT * l1_mean + SSIM_WEIGHT * (1 - ssim_mean)
  out[0] = acc[1] * inv + (1.0f - acc[0] * inv);
}

extern "C" void kernel_launch(void* const* d_in, const int* in_sizes, int n_in,
                              void* d_out, int out_size, void* d_ws, size_t ws_size,
                              hipStream_t stream) {
  const float* X = (const float*)d_in[0];   // outputs (img1)
  const float* Y = (const float*)d_in[1];   // labels  (img2)
  float* out = (float*)d_out;
  float* acc = (float*)d_ws;                // 2 floats of scratch

  zero_acc_kernel<<<1, 1, 0, stream>>>(acc);
  ssim_l1_kernel<<<TOTAL_TILES / 8, 256, 0, stream>>>(X, Y, acc);
  finalize_kernel<<<1, 1, 0, stream>>>(acc, out);
}
